// DGCNN_Encoder_73598559584322
// MI455X (gfx1250) — compile-verified, matrix operand order checked by reading, not run on hardware
//
#include <hip/hip_runtime.h>
#include <hip/hip_bf16.h>
#include <cstdint>
#include <cstddef>

// ---------------------------------------------------------------------------
// DGCNN encoder for MI455X (gfx1250).  Edge-MLP GEMMs run on
// v_wmma_f32_16x16x32_bf16 (wave32).  BN-over-edges + segment_max is fused
// via the monotone-affine identity: max(s*h+t) = s*max(h)+t for s>=0 (min
// otherwise), so only per-node max/min + per-channel sum/sumsq are kept.
// ---------------------------------------------------------------------------

typedef __attribute__((ext_vector_type(16))) __bf16 v16bf;
typedef __attribute__((ext_vector_type(8)))  float  v8f;

union AFrag { v16bf v; uint32_t d[8]; };

__device__ __forceinline__ uint16_t f2bf(float f) {
    uint32_t x = __float_as_uint(f);
    uint32_t r = (x + 0x7FFFu + ((x >> 16) & 1u)) >> 16;   // RNE
    return (uint16_t)r;
}

__device__ __forceinline__ void atomicMaxF(float* a, float v) {
    if (v >= 0.f) atomicMax((int*)a, __float_as_int(v));
    else          atomicMin((unsigned int*)a, __float_as_uint(v));
}
__device__ __forceinline__ void atomicMinF(float* a, float v) {
    if (v >= 0.f) atomicMin((int*)a, __float_as_int(v));
    else          atomicMax((unsigned int*)a, __float_as_uint(v));
}

// ---------------------------------------------------------------------------
// Small prep kernels
// ---------------------------------------------------------------------------

// out[n*Cin + k] = bf16(W[k*Cout + n])   (W row-major [Cin, Cout])
__global__ void transpose_bf16(const float* __restrict__ W, uint16_t* __restrict__ out,
                               int Cin, int Cout) {
    int i = blockIdx.x * blockDim.x + threadIdx.x;
    if (i >= Cin * Cout) return;
    int n = i / Cin, k = i - n * Cin;
    out[(size_t)n * Cin + k] = f2bf(W[(size_t)k * Cout + n]);
}

__global__ void init_bufs(float* __restrict__ nmax, float* __restrict__ nmin,
                          float* __restrict__ stats, int nelem, int nstats) {
    int i = blockIdx.x * blockDim.x + threadIdx.x;
    if (i < nelem) { nmax[i] = -__builtin_inff(); nmin[i] = __builtin_inff(); }
    if (i < nstats) stats[i] = 0.f;
}

// h = x @ W(3x64) + b
__global__ __launch_bounds__(256) void lin_forward(const float* __restrict__ x,
                                                   const float* __restrict__ w,
                                                   const float* __restrict__ b,
                                                   float* __restrict__ h, int N) {
    int i = blockIdx.x * blockDim.x + threadIdx.x;
    if (i >= N * 64) return;
    int n = i >> 6, c = i & 63;
    float acc = b[c];
#pragma unroll
    for (int k = 0; k < 3; ++k) acc += x[n * 3 + k] * w[k * 64 + c];
    h[i] = acc;
}

// ---------------------------------------------------------------------------
// Edge-MLP + stats + segment max/min.  64 edges per workgroup, 8 waves.
// CIN = node feature dim, CM = 2*CIN, CHID = hidden, COUT = 128.
// ---------------------------------------------------------------------------
template <int CIN, int CM, int CHID, int COUT>
__global__ __launch_bounds__(256) void edge_mlp_wmma(
        const float* __restrict__ hin,      // [N, CIN]
        const int*   __restrict__ ei,       // [2, E] (src row, dst row)
        const uint16_t* __restrict__ w1T,   // [CHID, CM] bf16
        const float* __restrict__ b1,       // [CHID]
        const uint16_t* __restrict__ w2T,   // [COUT, CHID] bf16
        const float* __restrict__ b2,       // [COUT]
        float* __restrict__ nmax,           // [N, COUT]  init -inf
        float* __restrict__ nmin,           // [N, COUT]  init +inf
        float* __restrict__ stats,          // [2*COUT]   sum, sumsq
        int N, int E, int Etot)
{
    __shared__ uint16_t mt[64 * CM];        // message tile, bf16
    __shared__ uint16_t h1[64 * CHID];      // hidden tile, bf16
    __shared__ int   edst[64];
    __shared__ float ssum[COUT];
    __shared__ float ssq[COUT];

    const int tid = threadIdx.x;
    const int e0  = blockIdx.x * 64;

    for (int c = tid; c < COUT; c += 256) { ssum[c] = 0.f; ssq[c] = 0.f; }

    // warm L2/L0 with weight panels (gfx1250 global_prefetch path)
    if (tid < 32) {
        __builtin_prefetch(w1T + (size_t)tid * ((size_t)CHID * CM / 32), 0, 3);
        __builtin_prefetch(w2T + (size_t)tid * ((size_t)COUT * CHID / 32), 0, 3);
    }

    // ---- build message tile: m = [xi, xj - xi], 4 threads per edge row ----
    {
        const int r = tid >> 2, sub = tid & 3;
        const int e = e0 + r;
        int s = -1, d = -1;
        if (e < E)         { s = ei[e]; d = ei[E + e]; }
        else if (e < Etot) { s = e - E; d = e - E; }          // self loops
        if (sub == 0) edst[r] = d;
        const float* hi = hin + (size_t)(d < 0 ? 0 : d) * CIN;
        const float* hj = hin + (size_t)(s < 0 ? 0 : s) * CIN;
        const int cw = CIN / 4;
        for (int c = sub * cw; c < sub * cw + cw; ++c) {
            float xi = (d >= 0) ? hi[c] : 0.f;
            float xj = (d >= 0) ? hj[c] : 0.f;
            mt[r * CM + c]       = f2bf(xi);
            mt[r * CM + CIN + c] = f2bf(xj - xi);
        }
    }
    __syncthreads();

    const int wave = tid >> 5;
    const int lane = tid & 31;
    const int hf   = lane >> 4;        // which 16-lane half
    const int l15  = lane & 15;

    // ---- GEMM1: [64,CM] @ [CM,CHID] -> relu -> bf16 in LDS ----------------
    for (int t = wave; t < 4 * (CHID / 16); t += 8) {
        const int m0 = (t & 3) << 4;
        const int n0 = (t >> 2) << 4;
        v8f acc = {0.f, 0.f, 0.f, 0.f, 0.f, 0.f, 0.f, 0.f};
        const uint32_t* arow = (const uint32_t*)(mt + (size_t)(m0 + l15) * CM);
        const uint32_t* brow = (const uint32_t*)(w1T + (size_t)(n0 + l15) * CM);
        for (int kb = 0; kb < CM; kb += 32) {
            AFrag a, b;
            const int d0 = (kb >> 1) + hf * 4;
#pragma unroll
            for (int i = 0; i < 4; ++i) { a.d[i] = arow[d0 + i]; a.d[4 + i] = arow[d0 + 8 + i]; }
#pragma unroll
            for (int i = 0; i < 4; ++i) { b.d[i] = brow[d0 + i]; b.d[4 + i] = brow[d0 + 8 + i]; }
            acc = __builtin_amdgcn_wmma_f32_16x16x32_bf16(false, a.v, false, b.v,
                                                          (short)0, acc, false, false);
        }
        const int ncol = n0 + l15;
        const float bias = b1[ncol];
#pragma unroll
        for (int r = 0; r < 8; ++r) {
            float v = acc[r] + bias;
            v = v > 0.f ? v : 0.f;
            h1[(size_t)(m0 + r + hf * 8) * CHID + ncol] = f2bf(v);
        }
    }
    __syncthreads();

    // ---- GEMM2: [64,CHID] @ [CHID,COUT] -> relu -> atomics ----------------
    for (int t = wave; t < 4 * (COUT / 16); t += 8) {
        const int m0 = (t & 3) << 4;
        const int n0 = (t >> 2) << 4;
        v8f acc = {0.f, 0.f, 0.f, 0.f, 0.f, 0.f, 0.f, 0.f};
        const uint32_t* arow = (const uint32_t*)(h1 + (size_t)(m0 + l15) * CHID);
        const uint32_t* brow = (const uint32_t*)(w2T + (size_t)(n0 + l15) * CHID);
        for (int kb = 0; kb < CHID; kb += 32) {
            AFrag a, b;
            const int d0 = (kb >> 1) + hf * 4;
#pragma unroll
            for (int i = 0; i < 4; ++i) { a.d[i] = arow[d0 + i]; a.d[4 + i] = arow[d0 + 8 + i]; }
#pragma unroll
            for (int i = 0; i < 4; ++i) { b.d[i] = brow[d0 + i]; b.d[4 + i] = brow[d0 + 8 + i]; }
            acc = __builtin_amdgcn_wmma_f32_16x16x32_bf16(false, a.v, false, b.v,
                                                          (short)0, acc, false, false);
        }
        const int ncol = n0 + l15;
        const float bias = b2[ncol];
#pragma unroll
        for (int r = 0; r < 8; ++r) {
            const int row = m0 + r + hf * 8;
            const int d = edst[row];
            if (d >= 0) {
                float v = acc[r] + bias;
                v = v > 0.f ? v : 0.f;
                atomicMaxF(&nmax[(size_t)d * COUT + ncol], v);
                atomicMinF(&nmin[(size_t)d * COUT + ncol], v);
                atomicAdd(&ssum[ncol], v);          // ds_add_f32
                atomicAdd(&ssq[ncol], v * v);
            }
        }
    }
    __syncthreads();

    for (int c = tid; c < COUT; c += 256) {
        atomicAdd(&stats[c], ssum[c]);
        atomicAdd(&stats[COUT + c], ssq[c]);
    }
}

// ---------------------------------------------------------------------------
// BN stat finalize (shared for edge-BN and node-BN):  scale/shift from
// sum/sumsq with biased variance, matching jnp var(0).
// ---------------------------------------------------------------------------
__global__ void bn_finalize(const float* __restrict__ stats,
                            const float* __restrict__ g, const float* __restrict__ b,
                            float cnt, float* __restrict__ scale, float* __restrict__ shift) {
    int c = threadIdx.x;
    if (c >= 128) return;
    float mean = stats[c] / cnt;
    float var  = stats[128 + c] / cnt - mean * mean;
    float sc   = g[c] * rsqrtf(var + 1e-5f);
    scale[c] = sc;
    shift[c] = b[c] - mean * sc;
}

// apply edge-BN through the segment max/min, accumulate node-BN stats,
// and re-arm the max/min buffers for the next layer.
__global__ __launch_bounds__(128) void conv_bn_pick(
        const float* __restrict__ nmax, const float* __restrict__ nmin,
        const float* __restrict__ scale, const float* __restrict__ shift,
        float* __restrict__ hout, float* __restrict__ nstats,
        float* __restrict__ nmaxw, float* __restrict__ nminw, int N) {
    const int c = threadIdx.x;                 // channel 0..127
    const float sc = scale[c], sh = shift[c];
    float ls = 0.f, lq = 0.f;
    const int r0 = blockIdx.x * 32;
    for (int r = r0; r < r0 + 32 && r < N; ++r) {
        size_t i = (size_t)r * 128 + c;
        float v = (sc >= 0.f) ? nmax[i] : nmin[i];
        float y = v * sc + sh;
        hout[i] = y;
        ls += y; lq += y * y;
        nmaxw[i] = -__builtin_inff();          // reset for next layer
        nminw[i] =  __builtin_inff();
    }
    atomicAdd(&nstats[c], ls);
    atomicAdd(&nstats[128 + c], lq);
}

__global__ void bn_apply(const float* __restrict__ hin,
                         const float* __restrict__ scale, const float* __restrict__ shift,
                         float* __restrict__ hout, int N, int relu) {
    int i = blockIdx.x * blockDim.x + threadIdx.x;
    if (i >= N * 128) return;
    int c = i & 127;
    float y = hin[i] * scale[c] + shift[c];
    if (relu) y = y > 0.f ? y : 0.f;
    hout[i] = y;
}

// ---------------------------------------------------------------------------
// Host launcher
// ---------------------------------------------------------------------------
extern "C" void kernel_launch(void* const* d_in, const int* in_sizes, int n_in,
                              void* d_out, int out_size, void* d_ws, size_t ws_size,
                              hipStream_t stream) {
    (void)n_in; (void)out_size; (void)ws_size;
    // inputs: dict insertion order  (x, edge_index, params{...})
    const float* x      = (const float*)d_in[0];
    const int*   ei     = (const int*)d_in[1];
    const float* lin_w  = (const float*)d_in[2];
    const float* lin_b  = (const float*)d_in[3];
    const float* c_w1[3] = {(const float*)d_in[4],  (const float*)d_in[10], (const float*)d_in[16]};
    const float* c_b1[3] = {(const float*)d_in[5],  (const float*)d_in[11], (const float*)d_in[17]};
    const float* c_w2[3] = {(const float*)d_in[6],  (const float*)d_in[12], (const float*)d_in[18]};
    const float* c_b2[3] = {(const float*)d_in[7],  (const float*)d_in[13], (const float*)d_in[19]};
    const float* c_bg[3] = {(const float*)d_in[8],  (const float*)d_in[14], (const float*)d_in[20]};
    const float* c_bb[3] = {(const float*)d_in[9],  (const float*)d_in[15], (const float*)d_in[21]};
    const float* o_bg[3] = {(const float*)d_in[22], (const float*)d_in[24], (const float*)d_in[26]};
    const float* o_bb[3] = {(const float*)d_in[23], (const float*)d_in[25], (const float*)d_in[27]};

    const int N    = in_sizes[0] / 3;
    const int E    = in_sizes[1] / 2;
    const int Etot = E + N;

    // workspace layout (~21 MB)
    float* bufA  = (float*)d_ws;
    float* bufB  = bufA + (size_t)N * 128;
    float* nmax  = bufB + (size_t)N * 128;
    float* nmin  = nmax + (size_t)N * 128;
    float* stats = nmin + (size_t)N * 128;   // conv[3]*256 then node[3]*256 floats
    float* sscl  = stats + 1536;             // per layer: cs(128) ct(128) ns(128) nt(128)
    uint16_t* w1T[3]; uint16_t* w2T[3];
    uint16_t* wp = (uint16_t*)(sscl + 3 * 512);
    const int w1ci[3] = {128, 256, 256}, w1co[3] = {256, 256, 128};
    const int w2ci[3] = {256, 256, 128}, w2co[3] = {128, 128, 128};
    for (int L = 0; L < 3; ++L) {
        w1T[L] = wp; wp += (size_t)w1ci[L] * w1co[L];
        w2T[L] = wp; wp += (size_t)w2ci[L] * w2co[L];
    }

    // prep: bf16 transposed weights
    for (int L = 0; L < 3; ++L) {
        int n1 = w1ci[L] * w1co[L], n2 = w2ci[L] * w2co[L];
        transpose_bf16<<<(n1 + 255) / 256, 256, 0, stream>>>(c_w1[L], w1T[L], w1ci[L], w1co[L]);
        transpose_bf16<<<(n2 + 255) / 256, 256, 0, stream>>>(c_w2[L], w2T[L], w2ci[L], w2co[L]);
    }
    init_bufs<<<(N * 128 + 255) / 256, 256, 0, stream>>>(nmax, nmin, stats, N * 128, 1536);
    lin_forward<<<(N * 64 + 255) / 256, 256, 0, stream>>>(x, lin_w, lin_b, bufA, N);

    const int eblk = (Etot + 63) / 64;
    for (int L = 0; L < 3; ++L) {
        float* cstat = stats + L * 256;
        float* nstat = stats + 768 + L * 256;
        float* cs = sscl + L * 512, *ct = cs + 128, *ns = ct + 128, *nt = ns + 128;

        if (L == 0)
            edge_mlp_wmma<64, 128, 256, 128><<<eblk, 256, 0, stream>>>(
                bufA, ei, w1T[0], c_b1[0], w2T[0], c_b2[0], nmax, nmin, cstat, N, E, Etot);
        else if (L == 1)
            edge_mlp_wmma<128, 256, 256, 128><<<eblk, 256, 0, stream>>>(
                bufA, ei, w1T[1], c_b1[1], w2T[1], c_b2[1], nmax, nmin, cstat, N, E, Etot);
        else
            edge_mlp_wmma<128, 256, 128, 128><<<eblk, 256, 0, stream>>>(
                bufA, ei, w1T[2], c_b1[2], w2T[2], c_b2[2], nmax, nmin, cstat, N, E, Etot);

        bn_finalize<<<1, 128, 0, stream>>>(cstat, c_bg[L], c_bb[L], (float)Etot, cs, ct);
        conv_bn_pick<<<(N + 31) / 32, 128, 0, stream>>>(nmax, nmin, cs, ct, bufB, nstat,
                                                        nmax, nmin, N);
        bn_finalize<<<1, 128, 0, stream>>>(nstat, o_bg[L], o_bb[L], (float)N, ns, nt);
        float* dst = (L == 2) ? (float*)d_out : bufA;
        bn_apply<<<(N * 128 + 255) / 256, 256, 0, stream>>>(bufB, ns, nt, dst, N, L != 2);
    }
}
